// RNN_echostate_32581621907699
// MI455X (gfx1250) — matrix operand order repaired and checked by reading, not compiled, not run on hardware
//
#include <hip/hip_runtime.h>

// Echo-state RNN, MI455X (gfx1250, wave32).
// Sequential recurrence (T=2048) => latency-bound => one persistent 512-thread
// workgroup (16 waves, 4/SIMD) on a single WGP.
//
// Capacity/traffic arithmetic:
//  - W_hh bf16 = 512 KB = entire WGP register file and > LDS spare space, so
//    it streams from L2 each step (512 KB/step, re-hit; 1 GB total), laid out
//    in WMMA-B-fragment order so each wave issues coalesced global_load_b128.
//    An opaque asm dependence keeps these loads inside the t-loop (otherwise
//    LICM hoists them into registers and spills to scratch -- seen in R1/R3).
//  - tanh(h) (32x512 bf16, 32 KB) in LDS is the A operand broadcast to all
//    waves: LDS traffic = waves * 32 KB/step. 16 waves (not 32) halves this
//    to 512 KB/step; each wave does 2 Mtiles x 2 Ntiles x 16 Ksteps = 64
//    v_wmma_f32_16x16x32_bf16, A fragments reused across Ntiles.
//  - 4 waves/SIMD doubles the VGPR cap (~256): live set ~150 regs, no spills.
//  - h state in VGPRs (16 threads/batch x 32 elems), matmul exchanged via a
//    small LDS acc buffer, outputs streamed to HBM with non-temporal stores.

#define NHID 512
#define NB   32
#define NT   2048
#define NWAVE 16
#define HTS  520   // ht row stride (bf16 units): 16B-aligned rows, bank-skewed
#define ACCS 516   // acc row stride (f32 units): 16B-aligned rows, bank-skewed

typedef __attribute__((ext_vector_type(16))) __bf16 v16bf;
typedef __attribute__((ext_vector_type(8)))  float  v8f;
typedef __attribute__((ext_vector_type(4)))  float  v4f;

__device__ __forceinline__ unsigned bf16rne(float f) {
  unsigned u = __float_as_uint(f);
  return (u + 0x7fffu + ((u >> 16) & 1u)) >> 16;
}
__device__ __forceinline__ float fast_tanh(float v) {
  // tanh(x) = 1 - 2/(exp(2x)+1); saturates correctly for |x| large.
  float e = __expf(2.0f * v);
  return 1.0f - 2.0f / (e + 1.0f);
}

// Pre-swizzle W_hh (fp32 row-major [n][k]) into bf16 B-fragment-major order
// for the 16-wave tiling.  Output halves:
//   o = ((((w*16 + kt)*2 + j)*2 + jj)*32 + lane)*8 + i
// holds W_hh[n][k] with n = 32w + 16j + (lane&15),
//                       k = 32kt + 8*(lane>>4) + 16jj + i.
// Per (wave,kt): 2 KB contiguous; per fragment piece: 512 B coalesced.
__global__ void whh_swizzle_bf16(const float* __restrict__ W,
                                 unsigned short* __restrict__ Wsw) {
  int o = blockIdx.x * blockDim.x + threadIdx.x;
  if (o >= NHID * NHID) return;
  const int i    = o & 7;
  const int lane = (o >> 3) & 31;
  const int jj   = (o >> 8) & 1;
  const int j    = (o >> 9) & 1;
  const int kt   = (o >> 10) & 15;
  const int w    = o >> 14;
  const int n = (w << 5) + (j << 4) + (lane & 15);
  const int k = (kt << 5) + ((lane >> 4) << 3) + (jj << 4) + i;
  Wsw[o] = (unsigned short)bf16rne(W[n * NHID + k]);
}

__global__ __launch_bounds__(512) void esn_persistent(
    const float* __restrict__ x, const float* __restrict__ h0,
    const float* __restrict__ W_ih, const unsigned short* __restrict__ Whh_sw,
    const float* __restrict__ W_hz, const float* __restrict__ W_zh,
    float* __restrict__ z_out, float* __restrict__ h_out) {
  extern __shared__ char smem[];
  unsigned short* ht = (unsigned short*)smem;                 // [NB][HTS] bf16 tanh(h)
  float* acc  = (float*)(smem + NB * HTS * sizeof(unsigned short)); // [NB][ACCS]
  float* zbuf = acc + NB * ACCS;                              // [2][NB] (parity)
  float* wihs = zbuf + 2 * NB;                                // [NHID]
  float* wzhs = wihs + NHID;
  float* whzs = wzhs + NHID;

  const int tid   = threadIdx.x;
  const int w     = tid >> 5;        // wave id (0..15)
  const int lane  = tid & 31;
  const int half  = lane >> 4;
  const int m16   = lane & 15;
  // matmul phase: wave w owns columns [32w, 32w+32): ntiles n0, n1
  const int ncol0 = (w << 5) + m16;
  // update phase: 16 threads per batch, 32 h-elements per thread
  const int ub = tid >> 4;           // batch (0..31)
  const int un = (tid & 15) << 5;    // first h index of this thread's slice

  for (int i = tid; i < NHID; i += 512) {
    wihs[i] = W_ih[i];
    wzhs[i] = W_zh[i];
    whzs[i] = W_hz[i];
  }

  // Per-wave streaming base for swizzled B fragments (uint4 units: 2048/wave).
  const uint4* wsw_base = (const uint4*)Whh_sw + (size_t)w * (16 * 2 * 2 * 32) + lane;

  __syncthreads();

  // ---- Prologue: h = h_0, publish tanh(h_0), z0 = tanh(h_0) @ W_hz^T
  float h[32];
  {
    const float4* hp = (const float4*)(h0 + (size_t)ub * NHID + un);
#pragma unroll
    for (int q = 0; q < 8; ++q) {
      float4 v = hp[q];
      h[4*q+0] = v.x; h[4*q+1] = v.y; h[4*q+2] = v.z; h[4*q+3] = v.w;
    }
    float th[32]; float s = 0.0f;
#pragma unroll
    for (int i = 0; i < 32; ++i) {
      th[i] = fast_tanh(h[i]);
      s += th[i] * whzs[un + i];
    }
    unsigned pk[16];
#pragma unroll
    for (int i = 0; i < 16; ++i)
      pk[i] = bf16rne(th[2*i]) | (bf16rne(th[2*i+1]) << 16);
    uint4* hts = (uint4*)(ht + ub * HTS + un);
#pragma unroll
    for (int q = 0; q < 4; ++q)
      hts[q] = make_uint4(pk[4*q+0], pk[4*q+1], pk[4*q+2], pk[4*q+3]);
#pragma unroll
    for (int off = 8; off >= 1; off >>= 1) s += __shfl_xor(s, off);
    if ((lane & 15) == 0) zbuf[ub] = s;  // z_{-1} into parity 0 (read at t=0)
  }
  __syncthreads();

  for (int t = 0; t < NT; ++t) {
    // ======== Phase 1: acc = tanh(h_prev) @ W_hh^T via WMMA ========
    // A (16x32 bf16, ISA layout) from LDS, reused for both N-tiles.
    // B streamed from L2; opaque asm keeps the loads inside the t-loop.
    const uint4* wp = wsw_base;
    asm volatile("" : "+v"(wp));  // defeat LICM: pointer is loop-variant now
    v8f d00 = {}, d01 = {}, d10 = {}, d11 = {};
    const unsigned short* a0r = ht + m16 * HTS;        // M-tile 0 (batches 0-15)
    const unsigned short* a1r = a0r + 16 * HTS;        // M-tile 1 (batches 16-31)
    const int klo = half * 8;
#pragma unroll
    for (int kt = 0; kt < 16; ++kt) {
      union { uint4 u[2]; v16bf v; } b0, b1;
      b0.u[0] = wp[((kt * 2 + 0) * 2 + 0) * 32];
      b0.u[1] = wp[((kt * 2 + 0) * 2 + 1) * 32];
      b1.u[0] = wp[((kt * 2 + 1) * 2 + 0) * 32];
      b1.u[1] = wp[((kt * 2 + 1) * 2 + 1) * 32];
      const int kb = kt * 32 + klo;
      union { uint4 u[2]; v16bf v; } a0, a1;
      a0.u[0] = *(const uint4*)(a0r + kb);
      a0.u[1] = *(const uint4*)(a0r + kb + 16);
      a1.u[0] = *(const uint4*)(a1r + kb);
      a1.u[1] = *(const uint4*)(a1r + kb + 16);
      d00 = __builtin_amdgcn_wmma_f32_16x16x32_bf16(false, a0.v, false,
              b0.v, (short)0, d00, false, false);
      d01 = __builtin_amdgcn_wmma_f32_16x16x32_bf16(false, a0.v, false,
              b1.v, (short)0, d01, false, false);
      d10 = __builtin_amdgcn_wmma_f32_16x16x32_bf16(false, a1.v, false,
              b0.v, (short)0, d10, false, false);
      d11 = __builtin_amdgcn_wmma_f32_16x16x32_bf16(false, a1.v, false,
              b1.v, (short)0, d11, false, false);
    }
    // D layout: VGPR r -> M = r + 8*half (+16 for M-tile 1), N = ncol0 (+16).
#pragma unroll
    for (int r = 0; r < 8; ++r) {
      const int m0 = (half * 8 + r) * ACCS;
      const int m1 = (16 + half * 8 + r) * ACCS;
      acc[m0 + ncol0]      = d00[r];
      acc[m0 + ncol0 + 16] = d01[r];
      acc[m1 + ncol0]      = d10[r];
      acc[m1 + ncol0 + 16] = d11[r];
    }
    __syncthreads();

    // ======== Phase 2: leaky update, 16 threads per batch ========
    const float xv    = x[ub * NT + t];
    const float zprev = zbuf[(t & 1) * NB + ub];
    float a[32];
    {
      const float4* ap = (const float4*)(acc + ub * ACCS + un);
#pragma unroll
      for (int q = 0; q < 8; ++q) {
        float4 v = ap[q];
        a[4*q+0] = v.x; a[4*q+1] = v.y; a[4*q+2] = v.z; a[4*q+3] = v.w;
      }
    }
    float th[32]; float s = 0.0f;
#pragma unroll
    for (int i = 0; i < 32; ++i) {
      float dh = -h[i] + a[i] + xv * wihs[un + i] + zprev * wzhs[un + i];
      h[i] += 0.1f * dh;
      th[i] = fast_tanh(h[i]);
      s += th[i] * whzs[un + i];
    }
    unsigned pk[16];
#pragma unroll
    for (int i = 0; i < 16; ++i)
      pk[i] = bf16rne(th[2*i]) | (bf16rne(th[2*i+1]) << 16);
    uint4* hts = (uint4*)(ht + ub * HTS + un);
#pragma unroll
    for (int q = 0; q < 4; ++q)
      hts[q] = make_uint4(pk[4*q+0], pk[4*q+1], pk[4*q+2], pk[4*q+3]);

    // stream h_t to HBM: [b][t][n], 2 KB contiguous per batch, non-temporal
    v4f* ho = (v4f*)(h_out + ((size_t)ub * NT + t) * NHID + un);
#pragma unroll
    for (int q = 0; q < 8; ++q) {
      v4f hv = { h[4*q+0], h[4*q+1], h[4*q+2], h[4*q+3] };
      __builtin_nontemporal_store(hv, ho + q);
    }

    // z_t = sum_n tanh(h_t)*W_hz[n] : half-wave reduction, no atomics
#pragma unroll
    for (int off = 8; off >= 1; off >>= 1) s += __shfl_xor(s, off);
    if ((lane & 15) == 0) {
      zbuf[((t + 1) & 1) * NB + ub] = s;                  // consumed at t+1
      __builtin_nontemporal_store(s, z_out + ub * NT + t); // output z[b][t]
    }
    __syncthreads();
  }
}

extern "C" void kernel_launch(void* const* d_in, const int* in_sizes, int n_in,
                              void* d_out, int out_size, void* d_ws, size_t ws_size,
                              hipStream_t stream) {
  (void)in_sizes; (void)n_in; (void)out_size; (void)ws_size;
  const float* x   = (const float*)d_in[0];  // [32,2048,1]
  const float* h0  = (const float*)d_in[1];  // [32,512]
  const float* Wih = (const float*)d_in[2];  // [512,1]
  const float* Whh = (const float*)d_in[3];  // [512,512]
  const float* Whz = (const float*)d_in[4];  // [1,512]
  const float* Wzh = (const float*)d_in[5];  // [512,1]

  float* z_out = (float*)d_out;                        // [32,2048]
  float* h_out = z_out + (size_t)NB * NT;              // [32,2048,512]
  unsigned short* Wb = (unsigned short*)d_ws;          // 512 KB swizzled bf16 W_hh

  whh_swizzle_bf16<<<(NHID * NHID + 255) / 256, 256, 0, stream>>>(Whh, Wb);

  const size_t smem = NB * HTS * sizeof(unsigned short)  // ht bf16
                    + (size_t)NB * ACCS * sizeof(float)  // acc
                    + 2 * NB * sizeof(float)             // zbuf
                    + 3 * NHID * sizeof(float);          // W_ih/W_zh/W_hz
  (void)hipFuncSetAttribute(reinterpret_cast<const void*>(esn_persistent),
                            hipFuncAttributeMaxDynamicSharedMemorySize,
                            (int)smem);
  esn_persistent<<<1, 512, smem, stream>>>(x, h0, Wih, Wb, Whz, Wzh,
                                           z_out, h_out);
}
